// MyTail_39067022525010
// MI455X (gfx1250) — compile-verified, matrix operand order checked by reading, not run on hardware
//
#include <hip/hip_runtime.h>

// ---- dims (fixed by the reference) ----
#define NFRAMES 64
#define CCH     1024
#define HHW     784      // 28*28
#define RR      64
#define DDIM    128
#define DFF     64
#define KQ      50176    // 1024*49
#define WIN     4
#define NIMG    16
#define EPSF    1e-6f

typedef __attribute__((ext_vector_type(16))) __bf16        v16bf;
typedef __attribute__((ext_vector_type(8)))  __bf16        v8bf;
typedef __attribute__((ext_vector_type(8)))  float         v8f;
typedef __attribute__((ext_vector_type(4)))  unsigned int  v4u;
typedef __attribute__((ext_vector_type(8)))  int           v8i;
typedef __attribute__((ext_vector_type(4)))  int           v4i;

#if __has_builtin(__builtin_amdgcn_tensor_load_to_lds) && \
    __has_builtin(__builtin_amdgcn_s_wait_tensorcnt)
#define USE_TDM 1
#endif

// ============================================================================
// Kernel 1: q = relu(roi_x[R, K] @ Wq[D, K]^T)   (K = C*7*7 = 50176)
// One block = one 16x16 output tile; 8 waves split K; LDS reduce + relu.
// ============================================================================
__global__ void qproj_kernel(const float* __restrict__ roi,
                             const float* __restrict__ Wq,
                             float* __restrict__ q) {
    const int tile = blockIdx.x;          // 0..31
    const int tr   = tile >> 3;           // M tile (R), 0..3
    const int td   = tile & 7;            // N tile (D), 0..7
    const int tid  = threadIdx.x;
    const int wave = tid >> 5;            // 0..7
    const int lane = tid & 31;
    const int half = lane >> 4;
    const int lrow = lane & 15;

    __shared__ float red[8][16][16];

    const int r  = tr * 16 + lrow;        // A row (per lane)
    const int dn = td * 16 + lrow;        // B column (per lane)
    const size_t rowA = (size_t)r  * KQ;
    const size_t rowB = (size_t)dn * KQ;

    v8f acc = {};
    const int kchunk = KQ / 8;            // 6272 (multiple of 32)
    const int k0 = wave * kchunk;
    for (int kb = k0; kb < k0 + kchunk; kb += 32) {
        // A fragment: lane half 0 -> K {0..7, 16..23}, half 1 -> K {8..15, 24..31}
        const float4* pa0 = (const float4*)(roi + rowA + kb + half * 8);
        const float4* pa1 = (const float4*)(roi + rowA + kb + 16 + half * 8);
        float4 a0 = pa0[0], a1 = pa0[1], a2 = pa1[0], a3 = pa1[1];
        const float4* pb0 = (const float4*)(Wq + rowB + kb + half * 8);
        const float4* pb1 = (const float4*)(Wq + rowB + kb + 16 + half * 8);
        float4 b0 = pb0[0], b1 = pb0[1], b2 = pb1[0], b3 = pb1[1];
        v16bf a, b;
        a[0]=(__bf16)a0.x;  a[1]=(__bf16)a0.y;  a[2]=(__bf16)a0.z;  a[3]=(__bf16)a0.w;
        a[4]=(__bf16)a1.x;  a[5]=(__bf16)a1.y;  a[6]=(__bf16)a1.z;  a[7]=(__bf16)a1.w;
        a[8]=(__bf16)a2.x;  a[9]=(__bf16)a2.y;  a[10]=(__bf16)a2.z; a[11]=(__bf16)a2.w;
        a[12]=(__bf16)a3.x; a[13]=(__bf16)a3.y; a[14]=(__bf16)a3.z; a[15]=(__bf16)a3.w;
        b[0]=(__bf16)b0.x;  b[1]=(__bf16)b0.y;  b[2]=(__bf16)b0.z;  b[3]=(__bf16)b0.w;
        b[4]=(__bf16)b1.x;  b[5]=(__bf16)b1.y;  b[6]=(__bf16)b1.z;  b[7]=(__bf16)b1.w;
        b[8]=(__bf16)b2.x;  b[9]=(__bf16)b2.y;  b[10]=(__bf16)b2.z; b[11]=(__bf16)b2.w;
        b[12]=(__bf16)b3.x; b[13]=(__bf16)b3.y; b[14]=(__bf16)b3.z; b[15]=(__bf16)b3.w;
        acc = __builtin_amdgcn_wmma_f32_16x16x32_bf16(
            false, a, false, b, (short)0, acc, false, false);
    }
    // D layout: VGPR j -> M = j + half*8 ; N = lane%16
    #pragma unroll
    for (int j = 0; j < 8; ++j)
        red[wave][half * 8 + j][lrow] = acc[j];
    __syncthreads();
    {
        const int m = tid >> 4, nn = tid & 15;
        float s = 0.f;
        #pragma unroll
        for (int w = 0; w < 8; ++w) s += red[w][m][nn];
        q[(size_t)(tr * 16 + m) * DDIM + td * 16 + nn] = fmaxf(s, 0.f);
    }
}

// ============================================================================
// Kernel 2: k/v projection. Per frame n:  k[n]^T[D, HW] = Wk^T[D,C] @ x[n][C,HW]
// Block = (p-tile, frame). x tile staged into LDS by the Tensor Data Mover
// (2D D#: 32 rows x 16 cols fp32, row stride 784, HW-padded to [32][17]);
// 8 waves (one d-tile each) accumulate k and v, sharing the x (B) fragment.
// ============================================================================
__global__ void kvproj_kernel(const float* __restrict__ x,
                              const float* __restrict__ Wk, const float* __restrict__ bk,
                              const float* __restrict__ Wv, const float* __restrict__ bv,
                              __bf16* __restrict__ kb16,
                              __bf16* __restrict__ vb16) {
    const int pt  = blockIdx.x;           // 0..48
    const int n   = blockIdx.y;           // frame
    const int p0  = pt * 16;
    const int tid = threadIdx.x;
    const int wave = tid >> 5;            // d-tile, 0..7
    const int lane = tid & 31;
    const int half = lane >> 4;
    const int lrow = lane & 15;

    __shared__ float xs[32][17];          // [c][p], padded (TDM pad 1 DW / 16 DW)

    const int d0 = wave * 16;
    const int dm = d0 + lrow;             // A row (d) for this lane

#ifdef USE_TDM
    const unsigned lds_base = (unsigned)(uintptr_t)(&xs[0][0]);
    const unsigned long long gbase =
        (unsigned long long)(uintptr_t)(x + (size_t)n * CCH * HHW + p0);
#endif

    v8f ck = {}, cv = {};
    for (int kc = 0; kc < CCH; kc += 32) {
        __syncthreads();                  // xs safe to overwrite
#ifdef USE_TDM
        if (wave == 0) {
            const unsigned long long ga = gbase + (unsigned long long)kc * (HHW * 4ull);
            // D# group0: count=1, lds_addr, 57-bit global addr, type=2
            v4u g0 = { 1u,
                       lds_base,
                       (unsigned)ga,
                       ((unsigned)(ga >> 32) & 0x01FFFFFFu) | 0x80000000u };
            // D# group1: data_size=4B, pad_enable, pad_interval=16DW, pad_amount=1DW,
            //            tensor_dim0=784, tensor_dim1=32, tile_dim0=16, tile_dim1=32,
            //            tensor_dim0_stride=784
            v8i g1 = { (int)0x00D20000,
                       (int)(784u << 16),
                       (int)(32u  << 16),
                       (int)(16u  << 16),
                       32,
                       784,
                       0, 0 };
            v4i gz4 = { 0, 0, 0, 0 };
            v8i gz8 = { 0, 0, 0, 0, 0, 0, 0, 0 };
            __builtin_amdgcn_tensor_load_to_lds(g0, g1, gz4, gz4, gz8, 0);
            __builtin_amdgcn_s_wait_tensorcnt(0);
        }
#else
        {
            const int c = tid >> 4;       // 0..15
            const int p = tid & 15;
            const size_t base = ((size_t)n * CCH + kc) * HHW + p0 + p;
            xs[c][p]      = x[base + (size_t)c * HHW];
            xs[c + 16][p] = x[base + (size_t)(c + 16) * HHW];
        }
#endif
        __syncthreads();
        v16bf ak, av, bx;
        #pragma unroll
        for (int e = 0; e < 16; ++e) {
            const int k = ((e >> 3) << 4) + half * 8 + (e & 7);
            const int c = kc + k;
            ak[e] = (__bf16)Wk[(size_t)c * DDIM + dm];
            av[e] = (__bf16)Wv[(size_t)c * DDIM + dm];
            bx[e] = (__bf16)xs[k][lrow];
        }
        ck = __builtin_amdgcn_wmma_f32_16x16x32_bf16(
            false, ak, false, bx, (short)0, ck, false, false);
        cv = __builtin_amdgcn_wmma_f32_16x16x32_bf16(
            false, av, false, bx, (short)0, cv, false, false);
    }
    // store D[m=d][n=p] -> [N, HW, D] bf16 (+bias); 8 consecutive d -> one b128
    const int p = p0 + lrow;
    v8bf sk, sv;
    #pragma unroll
    for (int j = 0; j < 8; ++j) {
        const int d = d0 + half * 8 + j;
        sk[j] = (__bf16)(ck[j] + bk[d]);
        sv[j] = (__bf16)(cv[j] + bv[d]);
    }
    const size_t off8 = ((size_t)n * HHW + p) * DDIM + d0 + half * 8;
    *(v8bf*)(kb16 + off8) = sk;
    *(v8bf*)(vb16 + off8) = sv;
}

// ============================================================================
// Kernel 3: one TX layer for all 64 queries (block r = one query row).
// scores -> softmax -> A -> LN1 -> FFN -> LN2, all fp32 (k/v read as bf16).
// ============================================================================
__global__ void tx_kernel(const float* __restrict__ qin, float* __restrict__ qout,
                          const __bf16* __restrict__ kb16,
                          const __bf16* __restrict__ vb16,
                          const int* __restrict__ cr,
                          const float* __restrict__ al1, const float* __restrict__ be1,
                          const float* __restrict__ al2, const float* __restrict__ be2,
                          const float* __restrict__ W1, const float* __restrict__ bb1,
                          const float* __restrict__ W2, const float* __restrict__ bb2) {
    const int r   = blockIdx.x;
    const int tid = threadIdx.x;          // 256 threads

    __shared__ float qs[DDIM];
    __shared__ float scores[WIN * HHW];   // 3136 floats
    __shared__ float red[256];
    __shared__ float accA[2][DDIM];
    __shared__ float qs2[DDIM];
    __shared__ float hs[DFF];

    if (tid < DDIM) qs[tid] = qin[(size_t)r * DDIM + tid];
    __syncthreads();
    const int start = cr[r * 3 + 1];
    const float inv_sqrt_d = 0.08838834764831845f;   // 1/sqrt(128)

    // ---- scores = (q . k) / sqrt(d), track max ----
    float lmax = -3.0e38f;
    for (int f = 0; f < WIN; ++f) {
        const size_t kbase = (size_t)(start + f) * HHW * DDIM;
        for (int p = tid; p < HHW; p += 256) {
            const v8bf* kp8 = (const v8bf*)(kb16 + kbase + (size_t)p * DDIM);
            __builtin_prefetch((const void*)(kb16 + kbase + (size_t)(p + 256) * DDIM), 0, 1);
            float s = 0.f;
            #pragma unroll
            for (int i = 0; i < 16; ++i) {
                const v8bf kv = kp8[i];
                #pragma unroll
                for (int j = 0; j < 8; ++j) s += qs[i * 8 + j] * (float)kv[j];
            }
            s *= inv_sqrt_d;
            scores[f * HHW + p] = s;
            lmax = fmaxf(lmax, s);
        }
    }
    red[tid] = lmax; __syncthreads();
    for (int o = 128; o > 0; o >>= 1) {
        if (tid < o) red[tid] = fmaxf(red[tid], red[tid + o]);
        __syncthreads();
    }
    const float mx = red[0];
    __syncthreads();

    // ---- exp and sum ----
    float lsum = 0.f;
    for (int s = tid; s < WIN * HHW; s += 256) {
        const float e = expf(scores[s] - mx);
        scores[s] = e; lsum += e;
    }
    red[tid] = lsum; __syncthreads();
    for (int o = 128; o > 0; o >>= 1) {
        if (tid < o) red[tid] += red[tid + o];
        __syncthreads();
    }
    const float inv_sum = 1.0f / red[0];
    __syncthreads();

    // ---- A[d] = sum_s p_s * v[s][d] (2-way split over s) ----
    {
        const int g = tid >> 7;
        const int d = tid & 127;
        float acc = 0.f;
        for (int f = 0; f < WIN; ++f) {
            const size_t vbase = (size_t)(start + f) * HHW * DDIM + d;
            for (int p = g; p < HHW; p += 2)
                acc += scores[f * HHW + p] * (float)vb16[vbase + (size_t)p * DDIM];
        }
        accA[g][d] = acc;
    }
    __syncthreads();

    // ---- LN1 on z = A + q  (torch std: ddof=1, denom = std + eps) ----
    float z = 0.f;
    if (tid < DDIM) z = (accA[0][tid] + accA[1][tid]) * inv_sum + qs[tid];
    red[tid] = (tid < DDIM) ? z : 0.f; __syncthreads();
    for (int o = 128; o > 0; o >>= 1) { if (tid < o) red[tid] += red[tid + o]; __syncthreads(); }
    const float mean1 = red[0] * (1.0f / DDIM);
    __syncthreads();
    float dv = (tid < DDIM) ? (z - mean1) : 0.f;
    red[tid] = dv * dv; __syncthreads();
    for (int o = 128; o > 0; o >>= 1) { if (tid < o) red[tid] += red[tid + o]; __syncthreads(); }
    const float std1 = sqrtf(red[0] * (1.0f / (DDIM - 1)));
    __syncthreads();
    if (tid < DDIM) qs2[tid] = al1[tid] * (z - mean1) / (std1 + EPSF) + be1[tid];
    __syncthreads();

    // ---- FFN: relu(q_ @ W1 + b1) @ W2 + b2 ----
    if (tid < DFF) {
        float h = bb1[tid];
        #pragma unroll 4
        for (int d = 0; d < DDIM; ++d) h += qs2[d] * W1[(size_t)d * DFF + tid];
        hs[tid] = fmaxf(h, 0.f);
    }
    __syncthreads();
    float z2 = 0.f;
    if (tid < DDIM) {
        float fv = bb2[tid];
        #pragma unroll 4
        for (int j = 0; j < DFF; ++j) fv += hs[j] * W2[(size_t)j * DDIM + tid];
        z2 = qs2[tid] + fv;
    }

    // ---- LN2 -> qout ----
    red[tid] = (tid < DDIM) ? z2 : 0.f; __syncthreads();
    for (int o = 128; o > 0; o >>= 1) { if (tid < o) red[tid] += red[tid + o]; __syncthreads(); }
    const float mean2 = red[0] * (1.0f / DDIM);
    __syncthreads();
    float dv2 = (tid < DDIM) ? (z2 - mean2) : 0.f;
    red[tid] = dv2 * dv2; __syncthreads();
    for (int o = 128; o > 0; o >>= 1) { if (tid < o) red[tid] += red[tid + o]; __syncthreads(); }
    const float std2 = sqrtf(red[0] * (1.0f / (DDIM - 1)));
    __syncthreads();
    if (tid < DDIM)
        qout[(size_t)r * DDIM + tid] = al2[tid] * (z2 - mean2) / (std2 + EPSF) + be2[tid];
}

// ============================================================================
// Kernel 4: head add + per-image mean over 4 ROIs -> out[16, 128]
// ============================================================================
__global__ void combine_kernel(const float* __restrict__ o0,
                               const float* __restrict__ o1,
                               float* __restrict__ out) {
    const int i = blockIdx.x;     // image
    const int d = threadIdx.x;    // 0..127
    float s = 0.f;
    #pragma unroll
    for (int g = 0; g < 4; ++g) {
        const size_t rr = (size_t)(i * 4 + g) * DDIM + d;
        s += o0[rr] + o1[rr];
    }
    out[(size_t)i * DDIM + d] = s * 0.25f;
}

// ============================================================================
extern "C" void kernel_launch(void* const* d_in, const int* in_sizes, int n_in,
                              void* d_out, int out_size, void* d_ws, size_t ws_size,
                              hipStream_t stream) {
    (void)in_sizes; (void)n_in; (void)out_size; (void)ws_size;
    const float* x      = (const float*)d_in[0];
    const float* roi    = (const float*)d_in[1];
    const int*   cr     = (const int*)d_in[2];
    const float* Wq     = (const float*)d_in[3];
    const float* Wk     = (const float*)d_in[4];
    const float* bk     = (const float*)d_in[5];
    const float* Wv     = (const float*)d_in[6];
    const float* bv     = (const float*)d_in[7];
    const float* alpha1 = (const float*)d_in[8];
    const float* beta1  = (const float*)d_in[9];
    const float* alpha2 = (const float*)d_in[10];
    const float* beta2  = (const float*)d_in[11];
    const float* W1     = (const float*)d_in[12];
    const float* b1     = (const float*)d_in[13];
    const float* W2     = (const float*)d_in[14];
    const float* b2     = (const float*)d_in[15];
    float* out = (float*)d_out;

    // workspace carve-up (floats first, then bf16 k/v; total ~25.9 MB)
    float* q0 = (float*)d_ws;             // [64,128] relu'd queries (shared by heads)
    float* qa = q0 + RR * DDIM;           // layer ping
    float* qb = qa + RR * DDIM;           // layer pong
    float* o0 = qb + RR * DDIM;           // head 0 result
    float* o1 = o0 + RR * DDIM;           // head 1 result
    __bf16* kb16 = (__bf16*)(o1 + RR * DDIM);
    __bf16* vb16 = kb16 + (size_t)NFRAMES * HHW * DDIM;

    qproj_kernel<<<32, 256, 0, stream>>>(roi, Wq, q0);
    kvproj_kernel<<<dim3(49, 64), 256, 0, stream>>>(x, Wk, bk, Wv, bv, kb16, vb16);

    for (int h = 0; h < 2; ++h) {
        float*       dst         = (h == 0) ? o0 : o1;
        const float* bufs_in[3]  = { q0, qa, qb };
        float*       bufs_out[3] = { qa, qb, dst };
        for (int j = 0; j < 3; ++j) {
            const int l = h * 3 + j;
            tx_kernel<<<64, 256, 0, stream>>>(
                bufs_in[j], bufs_out[j], kb16, vb16, cr,
                alpha1 + (size_t)l * DDIM, beta1 + (size_t)l * DDIM,
                alpha2 + (size_t)l * DDIM, beta2 + (size_t)l * DDIM,
                W1 + (size_t)l * DDIM * DFF, b1 + (size_t)l * DFF,
                W2 + (size_t)l * DFF * DDIM, b2 + (size_t)l * DDIM);
        }
    }
    combine_kernel<<<16, DDIM, 0, stream>>>(o0, o1, out);
}
